// GAT_43602507989241
// MI455X (gfx1250) — matrix operand order repaired and checked by reading, not compiled, not run on hardware
//
#include <hip/hip_runtime.h>

// ---------------------------------------------------------------------------
// GAT pipeline for MI455X (gfx1250, wave32, WMMA + async-to-LDS, dbl-buffer).
//   N=2048 nodes, D=512 feat, R=4 relations, 4 GAT layers.
// stage 1:  whT[r] = (X @ W[r])^T           v_wmma_f32_16x16x32_bf16
// stage A:  attbf[r] = softmax-masked attention matrix, bf16 (L2-resident)
// stage 2:  xr[r]  = elu( attbf[r] @ Wh[r] ) pure bf16 WMMA GEMM
// All LDS staging: contiguous 16B chunks via global_load_async_to_lds_b128,
// double-buffered; fragments read as ds_load_b128 (contiguous by layout).
// ---------------------------------------------------------------------------

constexpr int kN = 2048;
constexpr int kD = 512;
constexpr int kR = 4;
constexpr float kAlpha = 0.2f;
constexpr float kEps   = 1e-5f;
constexpr float kNeg   = -9.0e15f;

typedef __attribute__((ext_vector_type(16))) __bf16 v16bf;
typedef __attribute__((ext_vector_type(8)))  float  v8f;
typedef __attribute__((ext_vector_type(4)))  unsigned int v4u;
typedef __attribute__((ext_vector_type(4)))  float v4f;
typedef __attribute__((ext_vector_type(4)))  int v4i;

union FragBf { unsigned int u[8]; v4u q[2]; v16bf v; };
union PackBf { v4u q; __bf16 b[8]; };

__device__ __forceinline__ unsigned short bf16_bits(float f) {
  union { __bf16 b; unsigned short s; } c; c.b = (__bf16)f; return c.s;
}

// --- CDNA5 async copy: one 16B chunk global -> LDS, tracked by ASYNCcnt ----
__device__ __forceinline__ unsigned lds_off_of(const void* p) {
  return (unsigned)(unsigned long long)p;   // LDS aperture: addr[31:0] = offset
}
__device__ __forceinline__ void async_b128(unsigned lds_off, const void* g) {
  asm volatile("global_load_async_to_lds_b128 %0, %1, off"
               :: "v"(lds_off), "v"(g) : "memory");
}
__device__ __forceinline__ void wait_async0() {
  asm volatile("s_wait_asynccnt 0x0" ::: "memory");
}

// A-frag and B-frag: two contiguous 16B runs each (by layout)
__device__ __forceinline__ void load_frag_a(FragBf& a,
                                            const unsigned short* s,
                                            int ar, int h) {
  a.q[0] = *(const v4u*)(s + ar * 40 + 8 * h);
  a.q[1] = *(const v4u*)(s + ar * 40 + 16 + 8 * h);
}
__device__ __forceinline__ void load_frag_b(FragBf& b,
                                            const unsigned short* s,
                                            int bc, int h) {
  b.q[0] = *(const v4u*)(s + bc * 40 + 8 * h);
  b.q[1] = *(const v4u*)(s + bc * 40 + 16 + 8 * h);
}

// ---------------------------------------------------------------------------
__global__ __launch_bounds__(256) void cvt_bf16_kernel(
    const float* __restrict__ in, unsigned short* __restrict__ out, long n) {
  long i = (long)blockIdx.x * 256 + threadIdx.x;
  if (i < n) out[i] = bf16_bits(in[i]);
}

// wT[r][n][k] = bf16(W[r][k][n])   (pre-transpose so GEMM B-tiles are rowcopy)
__global__ __launch_bounds__(256) void cvt_wT_kernel(
    const float* __restrict__ W, unsigned short* __restrict__ wT) {
  long i = (long)blockIdx.x * 256 + threadIdx.x;
  if (i >= (long)kR * kD * kD) return;
  int r = (int)(i >> 18);           // D*D = 2^18
  int rem = (int)(i & ((1 << 18) - 1));
  int k = rem >> 9, n = rem & (kD - 1);
  wT[((long)r * kD + n) * kD + k] = bf16_bits(W[i]);
}

// ---------------------------------------------------------------------------
// Question attention pooling: qemb = softmax(qe1 @ wq) @ qe1
__global__ __launch_bounds__(256) void question_kernel(
    const float* __restrict__ qe, const float* __restrict__ wq,
    float* __restrict__ qemb) {
  __shared__ float sc[32];
  int tid = threadIdx.x, wave = tid >> 5, lane = tid & 31;
  for (int q = wave; q < 32; q += 8) {
    float s = 0.f;
    for (int d = lane; d < kD; d += 32) s += qe[q * kD + d] * wq[d];
    for (int off = 16; off > 0; off >>= 1) s += __shfl_xor(s, off, 32);
    if (lane == 0) sc[q] = s;
  }
  __syncthreads();
  if (tid == 0) {
    float m = sc[0];
    for (int q = 1; q < 32; ++q) m = fmaxf(m, sc[q]);
    float ssum = 0.f;
    for (int q = 0; q < 32; ++q) { sc[q] = __expf(sc[q] - m); ssum += sc[q]; }
    float inv = 1.f / ssum;
    for (int q = 0; q < 32; ++q) sc[q] *= inv;
  }
  __syncthreads();
  for (int d = tid; d < kD; d += 256) {
    float acc = 0.f;
    for (int q = 0; q < 32; ++q) acc += sc[q] * qe[q * kD + d];
    qemb[d] = acc;
  }
}

// u[i] = sum_j Wc[j,i]*qemb[j]; c0 = bc.qemb  ((x@Wc.T+bc)@qemb collapsed)
__global__ __launch_bounds__(256) void uvec_kernel(
    const float* __restrict__ Wc, const float* __restrict__ bc,
    const float* __restrict__ qemb, float* __restrict__ uvec,
    float* __restrict__ c0) {
  int i = blockIdx.x * 256 + threadIdx.x;
  float s = 0.f;
  for (int j = 0; j < kD; ++j) s += Wc[(long)j * kD + i] * qemb[j];
  uvec[i] = s;
  if (i == 0) {
    float c = 0.f;
    for (int j = 0; j < kD; ++j) c += bc[j] * qemb[j];
    c0[0] = c;
  }
}

__global__ __launch_bounds__(256) void gate_kernel(
    const float* __restrict__ x, const float* __restrict__ uvec,
    const float* __restrict__ c0, const float* __restrict__ qemb,
    float* __restrict__ xout) {
  __shared__ float red[256];
  int n = blockIdx.x, tid = threadIdx.x;
  const float* xrow = x + (long)n * kD;
  float v0 = xrow[tid], v1 = xrow[tid + 256];
  red[tid] = v0 * uvec[tid] + v1 * uvec[tid + 256];
  __syncthreads();
  for (int s = 128; s > 0; s >>= 1) {
    if (tid < s) red[tid] += red[tid + s];
    __syncthreads();
  }
  float p = 1.f / (1.f + __expf(-(red[0] + c0[0])));
  xout[(long)n * kD + tid]       = p * v0 + (1.f - p) * qemb[tid];
  xout[(long)n * kD + tid + 256] = p * v1 + (1.f - p) * qemb[tid + 256];
}

// ---------------------------------------------------------------------------
// Stage-1 GEMM: whT[r][d][m] = bf16( (X @ W[r])[m][d] ).  64x64 tile/block,
// 8 waves (4 along M x 2 along N), 2 WMMA/wave/K-step, double-buffered LDS.
__global__ __launch_bounds__(256) void gemm_xw_wmma(
    const unsigned short* __restrict__ xbf,    // [N,D] bf16
    const unsigned short* __restrict__ wbfT,   // [R,D(n),D(k)] bf16
    unsigned short* __restrict__ whT) {        // [R,D,N] bf16
  __shared__ __align__(16) unsigned short sA[2][64 * 40];
  __shared__ __align__(16) unsigned short sB[2][64 * 40];
  int r = blockIdx.z;
  int mbase = blockIdx.x * 64;
  int nbase = blockIdx.y * 64;
  int tid = threadIdx.x, wave = tid >> 5, lane = tid & 31;
  int wm = wave & 3, wd = wave >> 2;
  int h = lane >> 4, l16 = lane & 15;

  int srow = tid >> 2, scol = (tid & 3) * 8;   // 16B chunk per thread
  unsigned aoff0 = lds_off_of(&sA[0][srow * 40 + scol]);
  unsigned aoff1 = lds_off_of(&sA[1][srow * 40 + scol]);
  unsigned boff0 = lds_off_of(&sB[0][srow * 40 + scol]);
  unsigned boff1 = lds_off_of(&sB[1][srow * 40 + scol]);
  const unsigned short* ag = xbf + (long)(mbase + srow) * kD + scol;
  const unsigned short* bg = wbfT + ((long)r * kD + nbase + srow) * kD + scol;

  async_b128(aoff0, ag);
  async_b128(boff0, bg);

  int ar = wm * 16 + l16;
  int bc0 = wd * 32 + l16, bc1 = bc0 + 16;
  v8f acc0 = {}, acc1 = {};
  constexpr int T = kD / 32;
#pragma unroll 1
  for (int t = 0; t < T; ++t) {
    wait_async0();
    __syncthreads();
    if (t + 1 < T) {
      int kb = (t + 1) * 32;
      async_b128((t & 1) ? aoff0 : aoff1, ag + kb);
      async_b128((t & 1) ? boff0 : boff1, bg + kb);
    }
    const unsigned short* cA = sA[t & 1];
    const unsigned short* cB = sB[t & 1];
    FragBf a, b0, b1;
    load_frag_a(a, cA, ar, h);
    load_frag_b(b0, cB, bc0, h);
    load_frag_b(b1, cB, bc1, h);
    acc0 = __builtin_amdgcn_wmma_f32_16x16x32_bf16(
        false, a.v, false, b0.v, (short)0, acc0, false, false);
    acc1 = __builtin_amdgcn_wmma_f32_16x16x32_bf16(
        false, a.v, false, b1.v, (short)0, acc1, false, false);
  }
  // store transposed + bf16-packed: 8 consecutive m per lane -> one b128 store
  long mb = mbase + wm * 16 + 8 * h;       // C layout: VGPR i -> M = i + 8h
  int d0 = nbase + wd * 32 + l16, d1 = d0 + 16;
  PackBf o0, o1;
#pragma unroll
  for (int i = 0; i < 8; ++i) {
    o0.b[i] = (__bf16)acc0[i];
    o1.b[i] = (__bf16)acc1[i];
  }
  *(v4u*)(whT + ((long)r * kD + d0) * kN + mb) = o0.q;
  *(v4u*)(whT + ((long)r * kD + d1) * kN + mb) = o1.q;
}

// w1[r][i] = sum_o W[r][i][o]*a1[r][o]  (and w2 with a2): one wave per row
__global__ __launch_bounds__(256) void wvec_kernel(
    const float* __restrict__ W, const float* __restrict__ a1,
    const float* __restrict__ a2, float* __restrict__ w1,
    float* __restrict__ w2) {
  int wave = threadIdx.x >> 5, lane = threadIdx.x & 31;
  int row = blockIdx.x * 8 + wave;          // r*D + i
  int r = row >> 9;
  const float* wr = W + (long)row * kD;
  const float* a1r = a1 + r * kD;
  const float* a2r = a2 + r * kD;
  float s1 = 0.f, s2 = 0.f;
  for (int o = lane; o < kD; o += 32) {
    float v = wr[o];
    s1 += v * a1r[o];
    s2 += v * a2r[o];
  }
  for (int off = 16; off > 0; off >>= 1) {
    s1 += __shfl_xor(s1, off, 32);
    s2 += __shfl_xor(s2, off, 32);
  }
  if (lane == 0) { w1[row] = s1; w2[row] = s2; }
}

// f1[r,n] = x[n].w1[r]   (f1 = Wh.a1 = X.(W a1));  one wave per (r,n)
__global__ __launch_bounds__(256) void fdot_kernel(
    const float* __restrict__ x, const float* __restrict__ w1,
    const float* __restrict__ w2, float* __restrict__ f1,
    float* __restrict__ f2) {
  int wave = threadIdx.x >> 5, lane = threadIdx.x & 31;
  int row = blockIdx.x * 8 + wave;          // r*N + n
  int r = row >> 11, n = row & (kN - 1);
  const float* xr_ = x + (long)n * kD;
  const float* w1r = w1 + r * kD;
  const float* w2r = w2 + r * kD;
  float s1 = 0.f, s2 = 0.f;
  for (int d = lane; d < kD; d += 32) {
    float v = xr_[d];
    s1 += v * w1r[d];
    s2 += v * w2r[d];
  }
  for (int off = 16; off > 0; off >>= 1) {
    s1 += __shfl_xor(s1, off, 32);
    s2 += __shfl_xor(s2, off, 32);
  }
  if (lane == 0) { f1[row] = s1; f2[row] = s2; }
}

// Online masked softmax stats over m for each (r,n). One wave per row.
__global__ __launch_bounds__(256) void gat_softmax_stats(
    const int* __restrict__ adj, const float* __restrict__ f1,
    const float* __restrict__ f2, float* __restrict__ rmax,
    float* __restrict__ rsum) {
  int wave = threadIdx.x >> 5, lane = threadIdx.x & 31;
  int row = blockIdx.x * 8 + wave;
  int r = row >> 11;
  const int* arow = adj + (long)row * kN;
  const float* f2r = f2 + r * kN;
  float f1v = f1[row];
  float lm = -__builtin_inff(), ls = 0.f;
  for (int m = lane; m < kN; m += 32) {
    float t = f1v + f2r[m];
    t = t > 0.f ? t : kAlpha * t;
    float v = (arow[m] > 0) ? t : kNeg;
    if (v > lm) { ls *= __expf(lm - v); lm = v; }
    ls += __expf(v - lm);
  }
  for (int off = 16; off > 0; off >>= 1) {
    float om = __shfl_xor(lm, off, 32);
    float os = __shfl_xor(ls, off, 32);
    float nm = fmaxf(lm, om);
    ls = ls * __expf(lm - nm) + os * __expf(om - nm);
    lm = nm;
  }
  if (lane == 0) { rmax[row] = lm; rsum[row] = ls; }
}

// Materialize attention matrix in bf16: attbf[(r*N+n)*N+m]. One wave per row;
// each lane handles 8 consecutive m (b128 store), wave covers 256 m per iter.
__global__ __launch_bounds__(256) void att_mat_kernel(
    const int* __restrict__ adj, const float* __restrict__ f1,
    const float* __restrict__ f2, const float* __restrict__ rmax,
    const float* __restrict__ rsum, unsigned short* __restrict__ attbf) {
  int wave = threadIdx.x >> 5, lane = threadIdx.x & 31;
  int row = blockIdx.x * 8 + wave;          // r*N + n
  int r = row >> 11;
  const int* arow = adj + (long)row * kN;
  const float* f2r = f2 + r * kN;
  unsigned short* orow = attbf + (long)row * kN;
  float f1v = f1[row], rm = rmax[row], ri = 1.f / rsum[row];
#pragma unroll 1
  for (int it = 0; it < kN / 256; ++it) {
    int m0 = it * 256 + lane * 8;
    v4i a0 = *(const v4i*)(arow + m0);
    v4i a1 = *(const v4i*)(arow + m0 + 4);
    v4f q0 = *(const v4f*)(f2r + m0);
    v4f q1 = *(const v4f*)(f2r + m0 + 4);
    PackBf o;
#pragma unroll
    for (int j = 0; j < 8; ++j) {
      float t = f1v + (j < 4 ? q0[j] : q1[j - 4]);
      t = t > 0.f ? t : kAlpha * t;
      int av = (j < 4) ? a0[j] : a1[j - 4];
      float v = (av > 0) ? t : kNeg;
      o.b[j] = (__bf16)(__expf(v - rm) * ri);
    }
    *(v4u*)(orow + m0) = o.q;
  }
}

// ---------------------------------------------------------------------------
// Stage-2 GEMM: xr[r,n,:] = elu( attbf[r] @ Wh[r] ).  Same tiling as stage 1;
// A tiles rowcopied from attbf, B tiles from whT, double-buffered.
__global__ __launch_bounds__(256) void gat_attn_wmma(
    const unsigned short* __restrict__ attbf,  // [R*N, N] bf16
    const unsigned short* __restrict__ whT,    // [R, D, N] bf16
    float* __restrict__ xr) {                  // [R, N, D] f32
  __shared__ __align__(16) unsigned short sA[2][64 * 40];
  __shared__ __align__(16) unsigned short sB[2][64 * 40];
  int r = blockIdx.z;
  int mbase = blockIdx.x * 64;   // output node rows
  int dbase = blockIdx.y * 64;   // output feature cols
  long rb = (long)r * kN;
  int tid = threadIdx.x, wave = tid >> 5, lane = tid & 31;
  int wm = wave & 3, wd = wave >> 2;
  int h = lane >> 4, l16 = lane & 15;

  int srow = tid >> 2, scol = (tid & 3) * 8;
  unsigned aoff0 = lds_off_of(&sA[0][srow * 40 + scol]);
  unsigned aoff1 = lds_off_of(&sA[1][srow * 40 + scol]);
  unsigned boff0 = lds_off_of(&sB[0][srow * 40 + scol]);
  unsigned boff1 = lds_off_of(&sB[1][srow * 40 + scol]);
  const unsigned short* ag = attbf + (rb + mbase + srow) * (long)kN + scol;
  const unsigned short* bg = whT + ((long)r * kD + dbase + srow) * kN + scol;

  async_b128(aoff0, ag);
  async_b128(boff0, bg);

  int ar = wm * 16 + l16;
  int bc0 = wd * 32 + l16, bc1 = bc0 + 16;
  v8f acc0 = {}, acc1 = {};
  constexpr int T = kN / 32;
#pragma unroll 1
  for (int t = 0; t < T; ++t) {
    wait_async0();
    __syncthreads();
    if (t + 1 < T) {
      int kb = (t + 1) * 32;
      async_b128((t & 1) ? aoff0 : aoff1, ag + kb);
      async_b128((t & 1) ? boff0 : boff1, bg + kb);
    }
    const unsigned short* cA = sA[t & 1];
    const unsigned short* cB = sB[t & 1];
    FragBf a, b0, b1;
    load_frag_a(a, cA, ar, h);
    load_frag_b(b0, cB, bc0, h);
    load_frag_b(b1, cB, bc1, h);
    acc0 = __builtin_amdgcn_wmma_f32_16x16x32_bf16(
        false, a.v, false, b0.v, (short)0, acc0, false, false);
    acc1 = __builtin_amdgcn_wmma_f32_16x16x32_bf16(
        false, a.v, false, b1.v, (short)0, acc1, false, false);
  }
  int d0 = dbase + wd * 32 + l16, d1 = d0 + 16;
#pragma unroll
  for (int i = 0; i < 8; ++i) {
    int outm = mbase + wm * 16 + i + 8 * h;
    float v0 = acc0[i], v1 = acc1[i];
    v0 = v0 > 0.f ? v0 : (__expf(v0) - 1.f);       // elu
    v1 = v1 > 0.f ? v1 : (__expf(v1) - 1.f);
    xr[(rb + outm) * kD + d0] = v0;
    xr[(rb + outm) * kD + d1] = v1;
  }
}

// LayerNorm over last dim, then max over relations. One block per node.
__global__ __launch_bounds__(256) void lnorm_max_kernel(
    const float* __restrict__ xr, const float* __restrict__ g,
    const float* __restrict__ b, float* __restrict__ xout) {
  __shared__ float red[256];
  int n = blockIdx.x, tid = threadIdx.x;
  float g0 = g[tid], g1 = g[tid + 256];
  float b0 = b[tid], b1 = b[tid + 256];
  float o0 = -__builtin_inff(), o1 = -__builtin_inff();
  for (int r = 0; r < kR; ++r) {
    const float* xrow = xr + ((long)r * kN + n) * kD;
    float v0 = xrow[tid], v1 = xrow[tid + 256];
    red[tid] = v0 + v1;
    __syncthreads();
    for (int s = 128; s > 0; s >>= 1) {
      if (tid < s) red[tid] += red[tid + s];
      __syncthreads();
    }
    float mu = red[0] * (1.f / kD);
    __syncthreads();
    float d0 = v0 - mu, d1 = v1 - mu;
    red[tid] = d0 * d0 + d1 * d1;
    __syncthreads();
    for (int s = 128; s > 0; s >>= 1) {
      if (tid < s) red[tid] += red[tid + s];
      __syncthreads();
    }
    float rstd = rsqrtf(red[0] * (1.f / kD) + kEps);
    __syncthreads();
    o0 = fmaxf(o0, d0 * rstd * g0 + b0);
    o1 = fmaxf(o1, d1 * rstd * g1 + b1);
  }
  xout[(long)n * kD + tid]       = o0;
  xout[(long)n * kD + tid + 256] = o1;
}

__global__ __launch_bounds__(256) void max_rel_kernel(
    const float* __restrict__ xr, float* __restrict__ xout) {
  long i = (long)blockIdx.x * 256 + threadIdx.x;
  if (i >= (long)kN * kD) return;
  float m = xr[i];
  for (int r = 1; r < kR; ++r) m = fmaxf(m, xr[(long)r * kN * kD + i]);
  xout[i] = m;
}

// ---------------------------------------------------------------------------
// Choice head
__global__ __launch_bounds__(256) void meanq_kernel(
    const float* __restrict__ qe, float* __restrict__ meanq) {
  int d = blockIdx.x * 256 + threadIdx.x;
  float s = 0.f;
  for (int q = 0; q < 32; ++q) s += qe[q * kD + d];
  meanq[d] = s * (1.f / 32.f);
}

__global__ __launch_bounds__(256) void qproj_kernel(
    const float* __restrict__ Wqc, const float* __restrict__ bqc,
    const float* __restrict__ meanq, float* __restrict__ qproj) {
  int j = blockIdx.x * 256 + threadIdx.x;
  float s = 0.f;
  for (int i = 0; i < kD; ++i) s += meanq[i] * Wqc[(long)i * kD + j];
  qproj[j] = s + bqc[j];
}

__global__ __launch_bounds__(256) void choice_kernel(
    const float* __restrict__ xenc, const int* __restrict__ cidx,
    const float* __restrict__ qproj, const float* __restrict__ Wout,
    const float* __restrict__ bout, float* __restrict__ logits) {
  __shared__ float sc[64];
  __shared__ float red[256];
  int k = blockIdx.x, tid = threadIdx.x, wave = tid >> 5, lane = tid & 31;
  const int* idx = cidx + k * 64;
  for (int c = wave; c < 64; c += 8) {
    const float* xrow = xenc + (long)idx[c] * kD;
    float s = 0.f;
    for (int d = lane; d < kD; d += 32) s += xrow[d] * qproj[d];
    for (int off = 16; off > 0; off >>= 1) s += __shfl_xor(s, off, 32);
    if (lane == 0) sc[c] = s;
  }
  __syncthreads();
  if (tid == 0) {
    float m = sc[0];
    for (int c = 1; c < 64; ++c) m = fmaxf(m, sc[c]);
    float ssum = 0.f;
    for (int c = 0; c < 64; ++c) { sc[c] = __expf(sc[c] - m); ssum += sc[c]; }
    float inv = 1.f / ssum;
    for (int c = 0; c < 64; ++c) sc[c] *= inv;
  }
  __syncthreads();
  float part = 0.f;
  for (int d = tid; d < kD; d += 256) {
    float acc = 0.f;
    for (int c = 0; c < 64; ++c) acc += sc[c] * xenc[(long)idx[c] * kD + d];
    part += acc * Wout[d];
  }
  red[tid] = part;
  __syncthreads();
  for (int s = 128; s > 0; s >>= 1) {
    if (tid < s) red[tid] += red[tid + s];
    __syncthreads();
  }
  if (tid == 0) logits[k] = red[0] + bout[0];
}

__global__ void logsoftmax5_kernel(const float* __restrict__ logits,
                                   float* __restrict__ out) {
  if (threadIdx.x == 0) {
    float m = logits[0];
    for (int k = 1; k < 5; ++k) m = fmaxf(m, logits[k]);
    float s = 0.f;
    for (int k = 0; k < 5; ++k) s += __expf(logits[k] - m);
    float ls = logf(s) + m;
    for (int k = 0; k < 5; ++k) out[k] = logits[k] - ls;
  }
}

// ---------------------------------------------------------------------------
struct GatWs {
  unsigned short *xbf, *wbfT, *whT, *attbf;
  float *xr, *f1, *f2, *rmx, *rsm, *w1, *w2;
};

static void run_gat_layer(const float* xin, const float* Wl, const float* a1,
                          const float* a2, const int* adj, const float* lng,
                          const float* lnb, bool do_ln, float* xout,
                          const GatWs& ws, hipStream_t stream) {
  long nd = (long)kN * kD, wsz = (long)kR * kD * kD;
  cvt_bf16_kernel<<<(int)((nd + 255) / 256), 256, 0, stream>>>(xin, ws.xbf, nd);
  cvt_wT_kernel<<<(int)((wsz + 255) / 256), 256, 0, stream>>>(Wl, ws.wbfT);
  dim3 g1(kN / 64, kD / 64, kR);
  gemm_xw_wmma<<<g1, 256, 0, stream>>>(ws.xbf, ws.wbfT, ws.whT);
  wvec_kernel<<<kR * kD / 8, 256, 0, stream>>>(Wl, a1, a2, ws.w1, ws.w2);
  fdot_kernel<<<kR * kN / 8, 256, 0, stream>>>(xin, ws.w1, ws.w2, ws.f1, ws.f2);
  gat_softmax_stats<<<kR * kN / 8, 256, 0, stream>>>(adj, ws.f1, ws.f2,
                                                     ws.rmx, ws.rsm);
  att_mat_kernel<<<kR * kN / 8, 256, 0, stream>>>(adj, ws.f1, ws.f2, ws.rmx,
                                                  ws.rsm, ws.attbf);
  gat_attn_wmma<<<g1, 256, 0, stream>>>(ws.attbf, ws.whT, ws.xr);
  if (do_ln)
    lnorm_max_kernel<<<kN, 256, 0, stream>>>(ws.xr, lng, lnb, xout);
  else
    max_rel_kernel<<<(int)((nd + 255) / 256), 256, 0, stream>>>(ws.xr, xout);
}

extern "C" void kernel_launch(void* const* d_in, const int* in_sizes, int n_in,
                              void* d_out, int out_size, void* d_ws,
                              size_t ws_size, hipStream_t stream) {
  const float* x    = (const float*)d_in[0];
  const int*   adj  = (const int*)d_in[1];
  const float* qe1  = (const float*)d_in[2];
  // d_in[3] qnode_idx: unused by the reference
  const int*   cidx = (const int*)d_in[4];
  const float* Wc   = (const float*)d_in[5];
  const float* bc   = (const float*)d_in[6];
  const float* wq   = (const float*)d_in[7];
  const float* encW = (const float*)d_in[8];
  const float* ea1  = (const float*)d_in[9];
  const float* ea2  = (const float*)d_in[10];
  const float* decW = (const float*)d_in[11];
  const float* da1  = (const float*)d_in[12];
  const float* da2  = (const float*)d_in[13];
  const float* lng  = (const float*)d_in[14];
  const float* lnb  = (const float*)d_in[15];
  const float* Wqc  = (const float*)d_in[16];
  const float* bqc  = (const float*)d_in[17];
  const float* Wout = (const float*)d_in[18];
  const float* bout = (const float*)d_in[19];
  (void)in_sizes; (void)n_in; (void)out_size; (void)ws_size;

  char* p = (char*)d_ws;
  size_t off = 0;
  auto take = [&](size_t bytes) -> void* {
    void* r = p + off;
    off += (bytes + 255) & ~(size_t)255;
    return r;
  };
  float* bufA = (float*)take(sizeof(float) * (size_t)kN * kD);
  float* bufB = (float*)take(sizeof(float) * (size_t)kN * kD);
  GatWs ws;
  ws.xbf   = (unsigned short*)take(2ull * kN * kD);
  ws.wbfT  = (unsigned short*)take(2ull * kR * kD * kD);
  ws.whT   = (unsigned short*)take(2ull * kR * kN * kD);
  ws.attbf = (unsigned short*)take(2ull * kR * kN * kN);
  ws.xr    = (float*)take(4ull * kR * kN * kD);
  ws.f1    = (float*)take(4ull * kR * kN);
  ws.f2    = (float*)take(4ull * kR * kN);
  ws.rmx   = (float*)take(4ull * kR * kN);
  ws.rsm   = (float*)take(4ull * kR * kN);
  ws.w1    = (float*)take(4ull * kR * kD);
  ws.w2    = (float*)take(4ull * kR * kD);
  float* qemb   = (float*)take(4 * kD);
  float* uvec   = (float*)take(4 * kD);
  float* c0     = (float*)take(4);
  float* meanq  = (float*)take(4 * kD);
  float* qproj  = (float*)take(4 * kD);
  float* logits = (float*)take(4 * 8);

  question_kernel<<<1, 256, 0, stream>>>(qe1, wq, qemb);
  uvec_kernel<<<2, 256, 0, stream>>>(Wc, bc, qemb, uvec, c0);
  gate_kernel<<<kN, 256, 0, stream>>>(x, uvec, c0, qemb, bufA);

  const long wstride = (long)kR * kD * kD;
  const long astride = (long)kR * kD;
  run_gat_layer(bufA, encW,           ea1,           ea2,           adj,
                lng, lnb, true, bufB, ws, stream);
  run_gat_layer(bufB, encW + wstride, ea1 + astride, ea2 + astride, adj,
                lng, lnb, true, bufA, ws, stream);   // bufA = encoder output
  run_gat_layer(bufA, decW,           da1,           da2,           adj,
                lng, lnb, true, bufB, ws, stream);
  run_gat_layer(bufB, decW + wstride, da1 + astride, da2 + astride, adj,
                lng, lnb, false, (float*)d_out + 5, ws, stream);

  meanq_kernel<<<2, 256, 0, stream>>>(qe1, meanq);
  qproj_kernel<<<2, 256, 0, stream>>>(Wqc, bqc, meanq, qproj);
  choice_kernel<<<5, 256, 0, stream>>>(bufA, cidx, qproj, Wout, bout, logits);
  logsoftmax5_kernel<<<1, 32, 0, stream>>>(logits, (float*)d_out);
}
